// QuantizedLinear_25074019074595
// MI455X (gfx1250) — matrix operand order repaired and checked by reading, not compiled, not run on hardware
//
#include <hip/hip_runtime.h>

// ---------------------------------------------------------------------------
// QuantizedLinear: y[M,N] = scale[N] * (x[M,K] @ q[N,K]^T) + bias[N]
// M = 8192, K = 4096, N = 11008.
// Fast path: pre-convert W (int32->bf16, exact) and x (f32->bf16) into d_ws,
// then a bf16 WMMA GEMM whose K-stages are pure copies fed by CDNA5
// GLOBAL_LOAD_ASYNC_TO_LDS (ASYNCcnt) when available.
// Fallback (small ws): fused kernel with in-loop conversion (round-2 design).
// ---------------------------------------------------------------------------

#define IN_F    4096
#define OUT_F   11008
#define M_TOT   (4 * 2048)

typedef __attribute__((ext_vector_type(16))) __bf16 v16bf;
typedef __attribute__((ext_vector_type(8)))  __bf16 v8bf;
typedef __attribute__((ext_vector_type(4)))  __bf16 v4bf;
typedef __attribute__((ext_vector_type(8)))  float  v8f;
typedef __attribute__((ext_vector_type(4)))  float  v4f;
typedef __attribute__((ext_vector_type(8)))  int    v8i;
typedef __attribute__((ext_vector_type(4)))  int    v4i;

#if defined(__HIP_DEVICE_COMPILE__) && __has_builtin(__builtin_amdgcn_global_load_async_to_lds_b128)
#define HAS_ASYNC_LDS 1
#else
#define HAS_ASYNC_LDS 0
#endif

// ========================= pre-pass converters =============================

__global__ __launch_bounds__(256) void convert_w_bf16(
        const int* __restrict__ q, unsigned short* __restrict__ w) {
    const size_t i = ((size_t)blockIdx.x * 256 + threadIdx.x) * 8;
    v8i qi = *reinterpret_cast<const v8i*>(q + i);
    v8f qf = __builtin_convertvector(qi, v8f);     // exact for |q| <= 128
    *reinterpret_cast<v8bf*>(w + i) = __builtin_convertvector(qf, v8bf);
}

__global__ __launch_bounds__(256) void convert_x_bf16(
        const float* __restrict__ x, unsigned short* __restrict__ xb) {
    const size_t i = ((size_t)blockIdx.x * 256 + threadIdx.x) * 8;
    v8f xf = *reinterpret_cast<const v8f*>(x + i);
    *reinterpret_cast<v8bf*>(xb + i) = __builtin_convertvector(xf, v8bf);
}

// ========================= fragment loaders ================================
// A-matrix 16x32 bf16 fragment (ISA 7.12.2): lanes 0-15 hold K {0-7,16-23},
// lanes 16-31 hold K {8-15,24-31} of row (lane&15).
__device__ __forceinline__ v16bf load_frag_a(const unsigned short* base,
                                             int row, int hi, int stride) {
    union { v16bf v; uint4 q[2]; } u;
    const unsigned short* r = base + row * stride + hi * 8;
    u.q[0] = *reinterpret_cast<const uint4*>(r);
    u.q[1] = *reinterpret_cast<const uint4*>(r + 16);
    return u.v;
}
// B-matrix 32x16 bf16 fragment: lanes 0-15 hold K {0-7,16-23} of column (lane&15),
// lanes 16-31 hold K {8-15,24-31}.
__device__ __forceinline__ v16bf load_frag_b(const unsigned short* base,
                                             int row, int hi, int stride) {
    union { v16bf v; uint4 q[2]; } u;
    const unsigned short* r = base + row * stride + hi * 8;
    u.q[0] = *reinterpret_cast<const uint4*>(r);
    u.q[1] = *reinterpret_cast<const uint4*>(r + 16);
    return u.v;
}

// ========================= fast-path GEMM ==================================
// Block tile 256(M) x 128(N), BK=32. 8 waves in 4x2, each wave 64x64
// -> 4x4 fragments, 16 WMMA per stage per wave.
#define BM2   256
#define BN2   128
#define BK2   32
#define LDSS2 40      // BK2 + 8 pad halves; row = 80 B, 16B aligned

#if HAS_ASYNC_LDS
typedef __attribute__((address_space(1))) v4i gas_v4i;
typedef __attribute__((address_space(3))) v4i las_v4i;

__device__ __forceinline__ void cp16_async(const unsigned short* g, unsigned short* l) {
    __builtin_amdgcn_global_load_async_to_lds_b128(
        (gas_v4i*)(unsigned long long)g,
        (las_v4i*)(unsigned int)(unsigned long long)l,
        0, 0);
}

#if __has_builtin(__builtin_amdgcn_s_wait_asynccnt)
#define STAGE_WAIT() __builtin_amdgcn_s_wait_asynccnt(0)
#else
#define STAGE_WAIT() asm volatile("s_wait_asynccnt 0" ::: "memory")
#endif
#endif // HAS_ASYNC_LDS

struct Stage2Regs { uint4 a[4]; uint4 b[2]; };

__global__ __launch_bounds__(256) void qlinear_wmma_bf16_pre(
        const unsigned short* __restrict__ xb,   // [M, K] bf16
        const unsigned short* __restrict__ wb,   // [N, K] bf16
        const float* __restrict__ scales, const float* __restrict__ bias,
        float* __restrict__ out) {
    __shared__ __align__(16) unsigned short sA[2][BM2][LDSS2];   // 40 KB
    __shared__ __align__(16) unsigned short sB[2][BN2][LDSS2];   // 20 KB

    const int t      = threadIdx.x;
    const int lane   = t & 31;
    const int wave   = t >> 5;           // 0..7
    const int wm     = wave >> 1;        // 0..3 -> 64-row strip of M
    const int wn     = wave & 1;         // 0..1 -> 64-col strip of N
    const int lanelo = lane & 15;
    const int hi     = lane >> 4;

    const int n_base = blockIdx.x * BN2;
    const int m_base = blockIdx.y * BM2;

    // copy assignment: A: thread t copies row t (64 B); B: row t>>1, 32 B half
    const unsigned short* gA = xb + (size_t)(m_base + t) * IN_F;
    const int brow = t >> 1, bkh = (t & 1) * 16;
    const unsigned short* gB = wb + (size_t)(n_base + brow) * IN_F + bkh;

    v8f acc[4][4];
#pragma unroll
    for (int mf = 0; mf < 4; ++mf)
#pragma unroll
        for (int nf = 0; nf < 4; ++nf)
            acc[mf][nf] = (v8f){0.f, 0.f, 0.f, 0.f, 0.f, 0.f, 0.f, 0.f};

#if HAS_ASYNC_LDS
#define STAGE_ISSUE(k0, bufi)                                              \
    do {                                                                   \
        const unsigned short* ga_ = gA + (k0);                             \
        unsigned short* la_ = &sA[bufi][t][0];                             \
        cp16_async(ga_,      la_);                                         \
        cp16_async(ga_ +  8, la_ +  8);                                    \
        cp16_async(ga_ + 16, la_ + 16);                                    \
        cp16_async(ga_ + 24, la_ + 24);                                    \
        const unsigned short* gb_ = gB + (k0);                             \
        unsigned short* lb_ = &sB[bufi][brow][bkh];                        \
        cp16_async(gb_,     lb_);                                          \
        cp16_async(gb_ + 8, lb_ + 8);                                      \
    } while (0)

    STAGE_ISSUE(0, 0);
    STAGE_WAIT();
    __syncthreads();
#else
    Stage2Regs r;
#define STAGE_LOAD(k0)                                                     \
    do {                                                                   \
        const uint4* ga_ = reinterpret_cast<const uint4*>(gA + (k0));      \
        r.a[0] = ga_[0]; r.a[1] = ga_[1]; r.a[2] = ga_[2]; r.a[3] = ga_[3];\
        const uint4* gb_ = reinterpret_cast<const uint4*>(gB + (k0));      \
        r.b[0] = gb_[0]; r.b[1] = gb_[1];                                  \
    } while (0)
#define STAGE_STORE(bufi)                                                  \
    do {                                                                   \
        uint4* la_ = reinterpret_cast<uint4*>(&sA[bufi][t][0]);            \
        la_[0] = r.a[0]; la_[1] = r.a[1]; la_[2] = r.a[2]; la_[3] = r.a[3];\
        uint4* lb_ = reinterpret_cast<uint4*>(&sB[bufi][brow][bkh]);       \
        lb_[0] = r.b[0]; lb_[1] = r.b[1];                                  \
    } while (0)

    STAGE_LOAD(0);
    STAGE_STORE(0);
    __syncthreads();
#endif

    int buf = 0;
    for (int ks = 0; ks < IN_F / BK2; ++ks) {
        const bool more = (ks + 1) < (IN_F / BK2);
#if HAS_ASYNC_LDS
        if (more) STAGE_ISSUE((ks + 1) * BK2, buf ^ 1);
#else
        if (more) STAGE_LOAD((ks + 1) * BK2);
#endif

        v16bf bf[4];
#pragma unroll
        for (int nf = 0; nf < 4; ++nf)
            bf[nf] = load_frag_b(&sB[buf][0][0], wn * 64 + nf * 16 + lanelo, hi, LDSS2);
#pragma unroll
        for (int mf = 0; mf < 4; ++mf) {
            v16bf af = load_frag_a(&sA[buf][0][0], wm * 64 + mf * 16 + lanelo, hi, LDSS2);
#pragma unroll
            for (int nf = 0; nf < 4; ++nf)
                acc[mf][nf] = __builtin_amdgcn_wmma_f32_16x16x32_bf16(
                    false, af, false, bf[nf], (short)0, acc[mf][nf], false, false);
        }

#if HAS_ASYNC_LDS
        if (more) STAGE_WAIT();
#else
        if (more) STAGE_STORE(buf ^ 1);
#endif
        __syncthreads();
        buf ^= 1;
    }

    // Epilogue: C layout (ISA 7.12.2): VGPR v, lane l -> M = v+8*(l>>4), N = l&15.
    float sc[4], bi[4];
#pragma unroll
    for (int nf = 0; nf < 4; ++nf) {
        const int col = n_base + wn * 64 + nf * 16 + lanelo;
        sc[nf] = scales[col];
        bi[nf] = bias[col];
    }
#pragma unroll
    for (int mf = 0; mf < 4; ++mf) {
#pragma unroll
        for (int nf = 0; nf < 4; ++nf) {
            const int col = n_base + wn * 64 + nf * 16 + lanelo;
#pragma unroll
            for (int v = 0; v < 8; ++v) {
                const int row = m_base + wm * 64 + mf * 16 + hi * 8 + v;
                out[(size_t)row * OUT_F + col] = fmaf(acc[mf][nf][v], sc[nf], bi[nf]);
            }
        }
    }
}

// ========================= fallback fused GEMM (round-2) ===================
#define BM 128
#define BN 128
#define BK 32
#define LDSS 40

struct StageRegs { v4f a[4]; v4i b[4]; };

__device__ __forceinline__ void f_stage_load(StageRegs& r,
        const float* __restrict__ x, const int* __restrict__ qw,
        int m_base, int n_base, int k0, int t) {
    const int row = t >> 1, kh = (t & 1) * 16;
    const v4f* xa = reinterpret_cast<const v4f*>(x + (size_t)(m_base + row) * IN_F + k0 + kh);
    const v4i* qb = reinterpret_cast<const v4i*>(qw + (size_t)(n_base + row) * IN_F + k0 + kh);
#pragma unroll
    for (int i = 0; i < 4; ++i) { r.a[i] = xa[i]; r.b[i] = qb[i]; }
}

__device__ __forceinline__ void f_stage_store(const StageRegs& r,
        unsigned short* sA, unsigned short* sB, int t) {
    const int row = t >> 1, kh = (t & 1) * 16;
    v4bf a[4], b[4];
#pragma unroll
    for (int i = 0; i < 4; ++i) {
        a[i] = __builtin_convertvector(r.a[i], v4bf);
        v4f wf = __builtin_convertvector(r.b[i], v4f);
        b[i] = __builtin_convertvector(wf, v4bf);
    }
    v8bf* pa = reinterpret_cast<v8bf*>(sA + row * LDSS + kh);
    pa[0] = __builtin_shufflevector(a[0], a[1], 0, 1, 2, 3, 4, 5, 6, 7);
    pa[1] = __builtin_shufflevector(a[2], a[3], 0, 1, 2, 3, 4, 5, 6, 7);
    v8bf* pb = reinterpret_cast<v8bf*>(sB + row * LDSS + kh);
    pb[0] = __builtin_shufflevector(b[0], b[1], 0, 1, 2, 3, 4, 5, 6, 7);
    pb[1] = __builtin_shufflevector(b[2], b[3], 0, 1, 2, 3, 4, 5, 6, 7);
}

__global__ __launch_bounds__(256) void qlinear_wmma_bf16_fused(
        const float* __restrict__ x, const int* __restrict__ qw,
        const float* __restrict__ scales, const float* __restrict__ bias,
        float* __restrict__ out) {
    __shared__ __align__(16) unsigned short sA[2][BM][LDSS];
    __shared__ __align__(16) unsigned short sB[2][BN][LDSS];

    const int t = threadIdx.x, lane = t & 31, wave = t >> 5;
    const int wm = wave >> 2, wn = wave & 3;
    const int lanelo = lane & 15, hi = lane >> 4;
    const int n_base = blockIdx.x * BN, m_base = blockIdx.y * BM;

    v8f acc[4][2];
#pragma unroll
    for (int mf = 0; mf < 4; ++mf)
#pragma unroll
        for (int nf = 0; nf < 2; ++nf)
            acc[mf][nf] = (v8f){0.f, 0.f, 0.f, 0.f, 0.f, 0.f, 0.f, 0.f};

    StageRegs r;
    f_stage_load(r, x, qw, m_base, n_base, 0, t);
    f_stage_store(r, &sA[0][0][0], &sB[0][0][0], t);
    __syncthreads();

    int buf = 0;
    for (int ks = 0; ks < IN_F / BK; ++ks) {
        const bool more = (ks + 1) < (IN_F / BK);
        if (more) f_stage_load(r, x, qw, m_base, n_base, (ks + 1) * BK, t);

        v16bf bf[2];
#pragma unroll
        for (int nf = 0; nf < 2; ++nf)
            bf[nf] = load_frag_b(&sB[buf][0][0], wn * 32 + nf * 16 + lanelo, hi, LDSS);
#pragma unroll
        for (int mf = 0; mf < 4; ++mf) {
            v16bf af = load_frag_a(&sA[buf][0][0], wm * 64 + mf * 16 + lanelo, hi, LDSS);
#pragma unroll
            for (int nf = 0; nf < 2; ++nf)
                acc[mf][nf] = __builtin_amdgcn_wmma_f32_16x16x32_bf16(
                    false, af, false, bf[nf], (short)0, acc[mf][nf], false, false);
        }
        if (more) f_stage_store(r, &sA[buf ^ 1][0][0], &sB[buf ^ 1][0][0], t);
        __syncthreads();
        buf ^= 1;
    }

    const int c0 = n_base + wn * 32 + lanelo, c1 = c0 + 16;
    const float s0 = scales[c0], s1 = scales[c1];
    const float b0 = bias[c0],   b1 = bias[c1];
#pragma unroll
    for (int mf = 0; mf < 4; ++mf)
#pragma unroll
        for (int nf = 0; nf < 2; ++nf) {
            const int col = nf ? c1 : c0;
            const float ss = nf ? s1 : s0, bb = nf ? b1 : b0;
#pragma unroll
            for (int v = 0; v < 8; ++v) {
                const int row = m_base + wm * 64 + mf * 16 + hi * 8 + v;
                out[(size_t)row * OUT_F + col] = fmaf(acc[mf][nf][v], ss, bb);
            }
        }
}

// ========================= host launcher ===================================

extern "C" void kernel_launch(void* const* d_in, const int* in_sizes, int n_in,
                              void* d_out, int out_size, void* d_ws, size_t ws_size,
                              hipStream_t stream) {
    const float* x      = (const float*)d_in[0];
    const int*   qw     = (const int*)d_in[1];
    const float* scales = (const float*)d_in[2];
    const float* bias   = (const float*)d_in[3];
    float*       out    = (float*)d_out;

    const size_t W_EL = (size_t)OUT_F * IN_F;     // 45,088,768
    const size_t X_EL = (size_t)M_TOT * IN_F;     // 33,554,432
    const size_t need = (W_EL + X_EL) * sizeof(unsigned short);   // ~150 MB

    if (ws_size >= need) {
        unsigned short* wsW = (unsigned short*)d_ws;
        unsigned short* wsX = wsW + W_EL;
        convert_w_bf16<<<dim3((unsigned)(W_EL / (8 * 256))), dim3(256), 0, stream>>>(qw, wsW);
        convert_x_bf16<<<dim3((unsigned)(X_EL / (8 * 256))), dim3(256), 0, stream>>>(x, wsX);
        dim3 grid(OUT_F / BN2, M_TOT / BM2);      // 86 x 32
        qlinear_wmma_bf16_pre<<<grid, dim3(256), 0, stream>>>(wsX, wsW, scales, bias, out);
    } else {
        dim3 grid(OUT_F / BN, M_TOT / BM);        // 86 x 64
        qlinear_wmma_bf16_fused<<<grid, dim3(256), 0, stream>>>(x, qw, scales, bias, out);
    }
}
